// SpikingFieldUpdateNN_33904471834877
// MI455X (gfx1250) — matrix operand order repaired and checked by reading, not compiled, not run on hardware
//
#include <hip/hip_runtime.h>
#include <hip/hip_bf16.h>

// ---------------------------------------------------------------------------
// SNN forward: cur1 = x@W1^T+b1 (once); 10 steps of LIF -> GEMM -> LIF -> GEMM
// -> LIF -> spike accumulate; out = mean spikes. All GEMMs via bf16 WMMA.
// Block tile 128x128, 8 waves (4x2), wave tile 32x64 (2x4 WMMA accumulators).
// ---------------------------------------------------------------------------

typedef __attribute__((ext_vector_type(16))) __bf16 v16bf;
typedef __attribute__((ext_vector_type(8)))  __bf16 v8bf;
typedef __attribute__((ext_vector_type(8)))  float  v8f;

#define BETA   0.95f
#define THRESH 1.0f

// ---------------------------------------------------------------------------
// f32 -> bf16 conversion
// ---------------------------------------------------------------------------
__global__ __launch_bounds__(256) void cvt_f32_bf16(const float* __restrict__ in,
                                                    __bf16* __restrict__ out, int n) {
    int i = blockIdx.x * 256 + threadIdx.x;
    if (i < n) out[i] = (__bf16)in[i];
}

// ---------------------------------------------------------------------------
// LIF on precomputed current (layer 1): m1 <- beta*m1 + cur1 - reset; s1 = spike
// ---------------------------------------------------------------------------
__global__ __launch_bounds__(256) void lif1_kernel(const float* __restrict__ cur,
                                                   float* __restrict__ mem,
                                                   __bf16* __restrict__ spk, int n) {
    int i = blockIdx.x * 256 + threadIdx.x;
    if (i >= n) return;
    float mo    = mem[i];
    float reset = (mo - THRESH > 0.0f) ? THRESH : 0.0f;
    float mn    = BETA * mo + cur[i] - reset;
    mem[i]      = mn;
    spk[i]      = (__bf16)((mn - THRESH > 0.0f) ? 1.0f : 0.0f);
}

__global__ __launch_bounds__(256) void scale_kernel(float* __restrict__ out,
                                                    float s, int n) {
    int i = blockIdx.x * 256 + threadIdx.x;
    if (i < n) out[i] *= s;
}

// ---------------------------------------------------------------------------
// WMMA GEMM: C[M,N] = A[M,K](bf16,row-major) * B[N,K](bf16,row-major)^T + bias
// Block tile 128x128, 256 threads = 8 waves (wave32): 4 waves along M, 2 along
// N; wave tile 32 rows x 64 cols = 2x4 WMMA accumulators (8 WMMA per K-step).
// A tile staged via LDS; B fragments loaded directly (L2-resident weights).
// EPI: 0 = store f32 current; 1 = LIF -> spike bf16; 2 = LIF -> out += spike.
// Requires M%128==0, N%128==0, K%32==0.
// ---------------------------------------------------------------------------
template <int EPI>
__global__ __launch_bounds__(256) void gemm_wmma_bf16(
    const __bf16* __restrict__ A,
    const __bf16* __restrict__ B,
    const float*  __restrict__ bias,
    float*        __restrict__ Cout,    // EPI 0
    float*        __restrict__ mem,     // EPI 1/2: membrane state [M,N]
    __bf16*       __restrict__ Sout,    // EPI 1: spikes bf16 [M,N]
    float*        __restrict__ outAcc,  // EPI 2: spike accumulator [M,N]
    int M, int N, int K) {

    __shared__ __bf16 As[128 * 32];   // 8 KB A staging tile

    const int rowBase = blockIdx.y * 128;
    const int colBase = blockIdx.x * 128;

    const int lane = threadIdx.x & 31;
    const int wave = threadIdx.x >> 5;    // 0..7
    const int wm   = wave & 3;            // 4 waves along M (32 rows each)
    const int wn   = wave >> 2;           // 2 waves along N (64 cols each)
    const int hi16 = lane >> 4;           // half-wave select
    const int l16  = lane & 15;

    v8f acc[2][4] = {};

    // cooperative A-load indices: 256 threads cover 128 rows x 2 half-rows
    const int ldRow = threadIdx.x >> 1;
    const int ldCol = (threadIdx.x & 1) * 16;

    for (int k0 = 0; k0 < K; k0 += 32) {
        // ---- stage A tile (128x32 bf16) into LDS ----
        {
            const uint4* src = reinterpret_cast<const uint4*>(
                A + (size_t)(rowBase + ldRow) * K + k0 + ldCol);
            uint4* dst = reinterpret_cast<uint4*>(&As[ldRow * 32 + ldCol]);
            dst[0] = src[0];
            dst[1] = src[1];
        }
        __syncthreads();

        // ---- A fragments from LDS (16-bit A layout: lane half selects K-phase) ----
        v16bf aF[2];
#pragma unroll
        for (int ri = 0; ri < 2; ++ri) {
            const int r = wm * 32 + ri * 16 + l16;
            v8bf c0 = *reinterpret_cast<const v8bf*>(&As[r * 32 + hi16 * 8]);
            v8bf c1 = *reinterpret_cast<const v8bf*>(&As[r * 32 + 16 + hi16 * 8]);
#pragma unroll
            for (int i = 0; i < 8; ++i) { aF[ri][i] = c0[i]; aF[ri][i + 8] = c1[i]; }
        }

        // ---- B fragments direct from global (B row n == WMMA column n) ----
        v16bf bF[4];
#pragma unroll
        for (int ci = 0; ci < 4; ++ci) {
            const int n = colBase + wn * 64 + ci * 16 + l16;
            const __bf16* bp = B + (size_t)n * K + k0 + hi16 * 8;
            v8bf c0 = *reinterpret_cast<const v8bf*>(bp);
            v8bf c1 = *reinterpret_cast<const v8bf*>(bp + 16);
#pragma unroll
            for (int i = 0; i < 8; ++i) { bF[ci][i] = c0[i]; bF[ci][i + 8] = c1[i]; }
            __builtin_prefetch(bp + 32, 0, 3);   // next K-step, WGP-near prefetch
        }

        // ---- 8 WMMAs: 32x64 wave tile ----
#pragma unroll
        for (int ri = 0; ri < 2; ++ri)
#pragma unroll
            for (int ci = 0; ci < 4; ++ci)
                acc[ri][ci] = __builtin_amdgcn_wmma_f32_16x16x32_bf16(
                    /*neg_a=*/false, aF[ri], /*neg_b=*/false, bF[ci],
                    /*c_mod=*/(short)0, acc[ri][ci],
                    /*reuse_a=*/false, /*reuse_b=*/false);

        __syncthreads();
    }

    // ---- epilogue: C layout — VGPR i: M = i + 8*hi16, N = l16 ----
#pragma unroll
    for (int ri = 0; ri < 2; ++ri) {
#pragma unroll
        for (int ci = 0; ci < 4; ++ci) {
            const int col = colBase + wn * 64 + ci * 16 + l16;
            const float bv = bias[col];
#pragma unroll
            for (int i = 0; i < 8; ++i) {
                const int row = rowBase + wm * 32 + ri * 16 + hi16 * 8 + i;
                const size_t idx = (size_t)row * N + col;
                const float cur = acc[ri][ci][i] + bv;
                if constexpr (EPI == 0) {
                    Cout[idx] = cur;
                } else {
                    const float mo    = mem[idx];
                    const float reset = (mo - THRESH > 0.0f) ? THRESH : 0.0f;
                    const float mn    = BETA * mo + cur - reset;
                    mem[idx]          = mn;
                    const float spk   = (mn - THRESH > 0.0f) ? 1.0f : 0.0f;
                    if constexpr (EPI == 1) Sout[idx] = (__bf16)spk;
                    if constexpr (EPI == 2) outAcc[idx] += spk;
                }
            }
        }
    }
}

// ---------------------------------------------------------------------------
// Host launcher
// ---------------------------------------------------------------------------
extern "C" void kernel_launch(void* const* d_in, const int* in_sizes, int n_in,
                              void* d_out, int out_size, void* d_ws, size_t ws_size,
                              hipStream_t stream) {
    (void)n_in; (void)ws_size; (void)out_size;

    const float* x  = (const float*)d_in[0];
    const float* W1 = (const float*)d_in[1];
    const float* b1 = (const float*)d_in[2];
    const float* W2 = (const float*)d_in[3];
    const float* b2 = (const float*)d_in[4];
    const float* W3 = (const float*)d_in[5];
    const float* b3 = (const float*)d_in[6];
    float* out = (float*)d_out;

    const int H = in_sizes[2];              // 512
    const int D = in_sizes[6];              // 2048
    const int Bsz = in_sizes[0] / D;        // 8192
    const int NUM_STEPS = 10;

    // workspace carve-up (256B aligned)
    char* ws = (char*)d_ws;
    auto carve = [&](size_t bytes) {
        char* p = ws;
        ws += (bytes + 255) & ~(size_t)255;
        return p;
    };
    __bf16* xbf  = (__bf16*)carve((size_t)Bsz * D * 2);
    __bf16* w1bf = (__bf16*)carve((size_t)H * D * 2);
    __bf16* w2bf = (__bf16*)carve((size_t)H * H * 2);
    __bf16* w3bf = (__bf16*)carve((size_t)D * H * 2);
    float*  cur1 = (float*) carve((size_t)Bsz * H * 4);
    float*  m1   = (float*) carve((size_t)Bsz * H * 4);
    float*  m2   = (float*) carve((size_t)Bsz * H * 4);
    float*  m3   = (float*) carve((size_t)Bsz * D * 4);
    __bf16* s1   = (__bf16*)carve((size_t)Bsz * H * 2);
    __bf16* s2   = (__bf16*)carve((size_t)Bsz * H * 2);

    // convert inputs to bf16
    {
        int n = Bsz * D;
        cvt_f32_bf16<<<(n + 255) / 256, 256, 0, stream>>>(x, xbf, n);
        n = H * D;
        cvt_f32_bf16<<<(n + 255) / 256, 256, 0, stream>>>(W1, w1bf, n);
        n = H * H;
        cvt_f32_bf16<<<(n + 255) / 256, 256, 0, stream>>>(W2, w2bf, n);
        n = D * H;
        cvt_f32_bf16<<<(n + 255) / 256, 256, 0, stream>>>(W3, w3bf, n);
    }

    // zero membrane states and output accumulator (graph-capture safe)
    hipMemsetAsync(m1, 0, (size_t)Bsz * H * 4, stream);
    hipMemsetAsync(m2, 0, (size_t)Bsz * H * 4, stream);
    hipMemsetAsync(m3, 0, (size_t)Bsz * D * 4, stream);
    hipMemsetAsync(out, 0, (size_t)Bsz * D * 4, stream);

    dim3 blk(256);

    // GEMM1: cur1 = x @ W1^T + b1   [Bsz,H], K=D
    {
        dim3 grd(H / 128, Bsz / 128);
        gemm_wmma_bf16<0><<<grd, blk, 0, stream>>>(
            xbf, w1bf, b1, cur1, nullptr, nullptr, nullptr, Bsz, H, D);
    }

    const int n1 = Bsz * H;
    const int n3 = Bsz * D;
    for (int t = 0; t < NUM_STEPS; ++t) {
        // LIF layer 1 (current is constant across steps)
        lif1_kernel<<<(n1 + 255) / 256, 256, 0, stream>>>(cur1, m1, s1, n1);

        // layer 2: cur2 = s1 @ W2^T + b2, fused LIF -> s2   [Bsz,H], K=H
        {
            dim3 grd(H / 128, Bsz / 128);
            gemm_wmma_bf16<1><<<grd, blk, 0, stream>>>(
                s1, w2bf, b2, nullptr, m2, s2, nullptr, Bsz, H, H);
        }

        // layer 3: cur3 = s2 @ W3^T + b3, fused LIF -> out += spike  [Bsz,D], K=H
        {
            dim3 grd(D / 128, Bsz / 128);
            gemm_wmma_bf16<2><<<grd, blk, 0, stream>>>(
                s2, w3bf, b3, nullptr, m3, nullptr, out, Bsz, D, H);
        }
    }

    // mean over steps
    scale_kernel<<<(n3 + 255) / 256, 256, 0, stream>>>(out, 1.0f / NUM_STEPS, n3);
}
